// SmallDecoder_69243462746844
// MI455X (gfx1250) — compile-verified, matrix operand order checked by reading, not run on hardware
//
#include <hip/hip_runtime.h>
#include <hip/hip_bf16.h>
#include <stdint.h>

// ---------------------------------------------------------------------------
// Problem constants (from reference): B=4, S=2048, D=2048, H=16, K=128
// ---------------------------------------------------------------------------
#define BB 4
#define SS 2048
#define DD 2048
#define HH 16
#define MM (BB * SS)        // 8192 rows
#define CONCATD (DD + DD)   // 4096

typedef __attribute__((ext_vector_type(16))) __bf16 v16bf;
typedef __attribute__((ext_vector_type(8)))  float  v8f;

union FragBF {
    v16bf v;
    int4  q[2];
};

static __device__ __forceinline__ unsigned short f2bf(float f) {
    unsigned int u = __builtin_bit_cast(unsigned int, f);
    u += 0x7FFFu + ((u >> 16) & 1u);   // round-to-nearest-even
    return (unsigned short)(u >> 16);
}

// CDNA5 async copy global -> LDS (ASYNCcnt-tracked), 16B per lane.
static __device__ __forceinline__ void async_b128(unsigned lds_off, const void* gaddr) {
    asm volatile("global_load_async_to_lds_b128 %0, %1, off"
                 :: "v"(lds_off), "v"(gaddr)
                 : "memory");
}
static __device__ __forceinline__ void wait_async0() {
    asm volatile("s_wait_asynccnt 0x0" ::: "memory");
}

// ---------------------------------------------------------------------------
// Elementwise / small kernels
// ---------------------------------------------------------------------------

// x (f32 [M][D]) -> bf16 into concat[:, 0:D]  (concat row stride = CONCATD)
__global__ __launch_bounds__(256) void k_x_to_concat(const float* __restrict__ x,
                                                     unsigned short* __restrict__ concat) {
    size_t id = (size_t)blockIdx.x * 256 + threadIdx.x;
    if (id >= (size_t)MM * DD) return;
    size_t row = id / DD;
    int    col = (int)(id % DD);
    concat[row * CONCATD + col] = f2bf(x[id]);
}

// W [K][N] f32 -> Wt [N][K] bf16: LDS-tiled transpose, coalesced both sides.
__global__ __launch_bounds__(256) void k_cvt_transpose(const float* __restrict__ W,
                                                       unsigned short* __restrict__ Wt,
                                                       int Kd, int N) {
    __shared__ float tile[64][65];
    int tilesN = N >> 6;
    int tk = blockIdx.x / tilesN;
    int tn = blockIdx.x % tilesN;
    int k0 = tk << 6, n0 = tn << 6;
    int rr = threadIdx.x >> 6;    // 0..3
    int cc = threadIdx.x & 63;    // 0..63
    #pragma unroll
    for (int i = 0; i < 16; ++i) {
        int r = rr + i * 4;
        tile[r][cc] = W[(size_t)(k0 + r) * N + n0 + cc];
    }
    __syncthreads();
    #pragma unroll
    for (int i = 0; i < 16; ++i) {
        int r = rr + i * 4;   // n index
        Wt[(size_t)(n0 + r) * Kd + k0 + cc] = f2bf(tile[cc][r]);
    }
}

// logits[b,s,h] = sum_k x[b,s,k] * W_att[k,h]
__global__ __launch_bounds__(256) void k_logits(const float* __restrict__ x,
                                                const float* __restrict__ Watt,
                                                float* __restrict__ logits) {
    int id = blockIdx.x * 256 + threadIdx.x;   // [0, M*H)
    if (id >= MM * HH) return;
    int row = id >> 4;
    int h   = id & (HH - 1);
    const float* xr = x + (size_t)row * DD;
    float acc = 0.f;
    for (int k = 0; k < DD; ++k) acc += xr[k] * Watt[k * HH + h];
    logits[id] = acc;
}

// rowmax[b,h] = max_s logits[b,s,h]
__global__ __launch_bounds__(256) void k_rowmax(const float* __restrict__ logits,
                                                float* __restrict__ rowmax) {
    __shared__ float red[256];
    int b = blockIdx.x >> 4;
    int h = blockIdx.x & (HH - 1);
    float m = -3.402823466e38f;
    for (int s = threadIdx.x; s < SS; s += 256)
        m = fmaxf(m, logits[((size_t)b * SS + s) * HH + h]);
    red[threadIdx.x] = m;
    __syncthreads();
    for (int off = 128; off > 0; off >>= 1) {
        if ((int)threadIdx.x < off) red[threadIdx.x] = fmaxf(red[threadIdx.x], red[threadIdx.x + off]);
        __syncthreads();
    }
    if (threadIdx.x == 0) rowmax[blockIdx.x] = red[0];
}

// coef[b,s,h] = w / (w*(s+1) + 1e-30), w = exp((logit - max)*temp[h])
__global__ __launch_bounds__(256) void k_coef(const float* __restrict__ logits,
                                              const float* __restrict__ rowmax,
                                              const float* __restrict__ temp,
                                              float* __restrict__ coef) {
    int id = blockIdx.x * 256 + threadIdx.x;   // [0, M*H)
    if (id >= MM * HH) return;
    int h = id & (HH - 1);
    int s = (id >> 4) & (SS - 1);
    int b = id / (SS * HH);
    float w = __expf((logits[id] - rowmax[b * HH + h]) * temp[h]);
    coef[id] = w / (w * (float)(s + 1) + 1e-30f);
}

// ---------------------------------------------------------------------------
// 3-phase causal prefix sum over S (chunked for parallelism), fused scaling.
// ---------------------------------------------------------------------------
#define SCHUNK 128
#define NCHUNK (SS / SCHUNK)   // 16

// phase 1: partial[b,ch,c] = sum_{s in chunk} v[b,s,c]
__global__ __launch_bounds__(256) void k_scan_part(const float* __restrict__ v,
                                                   float* __restrict__ partial) {
    int id = blockIdx.x * 256 + threadIdx.x;   // [0, B*NCHUNK*D)
    if (id >= BB * NCHUNK * DD) return;
    int c  = id & (DD - 1);
    int ch = (id / DD) & (NCHUNK - 1);
    int b  = id / (DD * NCHUNK);
    size_t base = ((size_t)b * SS + ch * SCHUNK) * DD + c;
    float s = 0.f;
    for (int i = 0; i < SCHUNK; ++i) s += v[base + (size_t)i * DD];
    partial[id] = s;
}

// phase 2: exclusive scan over chunks per (b,c), in place
__global__ __launch_bounds__(256) void k_scan_offsets(float* __restrict__ partial) {
    int id = blockIdx.x * 256 + threadIdx.x;   // [0, B*D)
    if (id >= BB * DD) return;
    int c = id & (DD - 1);
    int b = id / DD;
    float run = 0.f;
    for (int ch = 0; ch < NCHUNK; ++ch) {
        size_t idx = ((size_t)b * NCHUNK + ch) * DD + c;
        float t = partial[idx];
        partial[idx] = run;
        run += t;
    }
}

// phase 3: pooled[b,s,c] = bf16( (offset + local cumsum) * coef[b,s,h] )
__global__ __launch_bounds__(256) void k_scan_apply(const float* __restrict__ v,
                                                    const float* __restrict__ partial,
                                                    const float* __restrict__ coef,
                                                    unsigned short* __restrict__ pooled) {
    int id = blockIdx.x * 256 + threadIdx.x;   // [0, B*NCHUNK*D)
    if (id >= BB * NCHUNK * DD) return;
    int c  = id & (DD - 1);
    int ch = (id / DD) & (NCHUNK - 1);
    int b  = id / (DD * NCHUNK);
    int h  = c >> 7;
    float run = partial[id];
    size_t base = ((size_t)b * SS + ch * SCHUNK) * DD + c;
    for (int i = 0; i < SCHUNK; ++i) {
        int s = ch * SCHUNK + i;
        run += v[base + (size_t)i * DD];
        pooled[base + (size_t)i * DD] = f2bf(run * coef[((size_t)b * SS + s) * HH + h]);
    }
}

// in-place layernorm over last dim of out [M][D]
__global__ __launch_bounds__(256) void k_layernorm(float* __restrict__ out,
                                                   const float* __restrict__ gamma,
                                                   const float* __restrict__ beta) {
    __shared__ float red[256];
    float* r = out + (size_t)blockIdx.x * DD;
    float vals[DD / 256];
    float sum = 0.f;
    #pragma unroll
    for (int i = 0; i < DD / 256; ++i) {
        vals[i] = r[threadIdx.x + i * 256];
        sum += vals[i];
    }
    red[threadIdx.x] = sum;
    __syncthreads();
    for (int off = 128; off > 0; off >>= 1) {
        if ((int)threadIdx.x < off) red[threadIdx.x] += red[threadIdx.x + off];
        __syncthreads();
    }
    float mean = red[0] * (1.0f / DD);
    __syncthreads();
    float ss = 0.f;
    #pragma unroll
    for (int i = 0; i < DD / 256; ++i) {
        float cdev = vals[i] - mean;
        ss += cdev * cdev;
    }
    red[threadIdx.x] = ss;
    __syncthreads();
    for (int off = 128; off > 0; off >>= 1) {
        if ((int)threadIdx.x < off) red[threadIdx.x] += red[threadIdx.x + off];
        __syncthreads();
    }
    float inv = rsqrtf(red[0] * (1.0f / DD) + 1e-6f);
    #pragma unroll
    for (int i = 0; i < DD / 256; ++i) {
        int c = threadIdx.x + i * 256;
        r[c] = (vals[i] - mean) * inv * gamma[c] + beta[c];
    }
}

// ---------------------------------------------------------------------------
// bf16 WMMA GEMM: C[M][N] = epilogue( A[M][Kdim] @ Bt[N][Kdim]^T )
// Block = 256 threads (8 waves), block tile 128x128, K-step 32.
// Double-buffered LDS staged with CDNA5 global_load_async_to_lds_b128
// (ASYNCcnt), one barrier per K-step.
// Wave grid 4(M) x 2(N); each wave owns a 32x64 tile = 2x4 WMMA 16x16 tiles.
// ---------------------------------------------------------------------------
#define EPI_F32        0   // store f32
#define EPI_BVEC_BF16  1   // + bias[(row/S)*bstride + col], store bf16
#define EPI_BRELU_BF16 2   // + bias[col], relu, store bf16
#define EPI_BRELU_RES  3   // + bias[col], relu, + resid[row*ldres+col], store f32

#define LDSW 40                 // padded row stride (elems); 80B rows, 16B aligned
#define BUFBYTES (128 * LDSW * 2)

template <int EPI>
__global__ __launch_bounds__(256) void gemm_bf16_wmma(
    const unsigned short* __restrict__ A, int lda,
    const unsigned short* __restrict__ Bt,
    void* __restrict__ Cout, int ldc,
    int Kdim,
    const float* __restrict__ bias, int bstride,
    const float* __restrict__ resid, int ldres,
    int Srows) {
    __shared__ __align__(16) unsigned short ldsA[2][128 * LDSW];
    __shared__ __align__(16) unsigned short ldsB[2][128 * LDSW];

    const int bm = blockIdx.y * 128;
    const int bn = blockIdx.x * 128;
    const int t     = threadIdx.x;
    const int lane  = t & 31;
    const int wid   = t >> 5;
    const int waveM = wid & 3;    // 0..3 -> 32-row slab
    const int waveN = wid >> 2;   // 0..1 -> 64-col slab
    const int lhalf = lane >> 4;  // 0/1
    const int l16   = lane & 15;
    const int koff  = lhalf * 8;

    v8f acc[2][4] = {};

    // staging assignment: thread covers rows lr and lr+64, 8 bf16 (16B) at lc
    const int lr = t >> 2;        // 0..63
    const int lc = (t & 3) * 8;   // 0,8,16,24

    const unsigned short* gA0 = A  + (size_t)(bm + lr)      * lda  + lc;
    const unsigned short* gA1 = A  + (size_t)(bm + lr + 64) * lda  + lc;
    const unsigned short* gB0 = Bt + (size_t)(bn + lr)      * (size_t)Kdim + lc;
    const unsigned short* gB1 = Bt + (size_t)(bn + lr + 64) * (size_t)Kdim + lc;

    const unsigned ldsA_base = (unsigned)(uintptr_t)&ldsA[0][0];
    const unsigned ldsB_base = (unsigned)(uintptr_t)&ldsB[0][0];
    const unsigned off0 = (unsigned)((lr * LDSW + lc) * 2);
    const unsigned off1 = (unsigned)(((lr + 64) * LDSW + lc) * 2);

    auto issue = [&](int buf, int ko) {
        unsigned ab = ldsA_base + buf * BUFBYTES;
        unsigned bb = ldsB_base + buf * BUFBYTES;
        async_b128(ab + off0, gA0 + ko);
        async_b128(ab + off1, gA1 + ko);
        async_b128(bb + off0, gB0 + ko);
        async_b128(bb + off1, gB1 + ko);
    };

    int cur = 0;
    issue(0, 0);

    for (int ko = 0; ko < Kdim; ko += 32) {
        wait_async0();
        __syncthreads();   // buf[cur] ready; buf[cur^1] no longer being read
        if (ko + 32 < Kdim) issue(cur ^ 1, ko + 32);

        const unsigned short* bufA = &ldsA[cur][0];
        const unsigned short* bufB = &ldsB[cur][0];

        FragBF fb[4];
        #pragma unroll
        for (int wn = 0; wn < 4; ++wn) {
            int n = waveN * 64 + wn * 16 + l16;
            fb[wn].q[0] = *(const int4*)&bufB[n * LDSW + koff];
            fb[wn].q[1] = *(const int4*)&bufB[n * LDSW + 16 + koff];
        }
        #pragma unroll
        for (int wm = 0; wm < 2; ++wm) {
            int m = waveM * 32 + wm * 16 + l16;
            FragBF fa;
            fa.q[0] = *(const int4*)&bufA[m * LDSW + koff];
            fa.q[1] = *(const int4*)&bufA[m * LDSW + 16 + koff];
            #pragma unroll
            for (int wn = 0; wn < 4; ++wn) {
                acc[wm][wn] = __builtin_amdgcn_wmma_f32_16x16x32_bf16(
                    false, fa.v, false, fb[wn].v,
                    (short)0, acc[wm][wn], false, false);
            }
        }
        cur ^= 1;
    }

    // Epilogue: D layout (16x16 f32): VGPR r -> M = r + 8*(lane>>4), N = lane&15
    #pragma unroll
    for (int wm = 0; wm < 2; ++wm) {
        #pragma unroll
        for (int wn = 0; wn < 4; ++wn) {
            #pragma unroll
            for (int r = 0; r < 8; ++r) {
                int row = bm + waveM * 32 + wm * 16 + lhalf * 8 + r;
                int col = bn + waveN * 64 + wn * 16 + l16;
                float v = acc[wm][wn][r];
                if (EPI == EPI_F32) {
                    ((float*)Cout)[(size_t)row * ldc + col] = v;
                } else if (EPI == EPI_BVEC_BF16) {
                    int b = row / Srows;
                    float o = v + bias[(size_t)b * bstride + col];
                    ((unsigned short*)Cout)[(size_t)row * ldc + col] = f2bf(o);
                } else if (EPI == EPI_BRELU_BF16) {
                    float o = fmaxf(v + bias[col], 0.f);
                    ((unsigned short*)Cout)[(size_t)row * ldc + col] = f2bf(o);
                } else {
                    float o = fmaxf(v + bias[col], 0.f);
                    o += resid[(size_t)row * ldres + col];
                    ((float*)Cout)[(size_t)row * ldc + col] = o;
                }
            }
        }
    }
}

// ---------------------------------------------------------------------------
// Host launcher
// ---------------------------------------------------------------------------
extern "C" void kernel_launch(void* const* d_in, const int* in_sizes, int n_in,
                              void* d_out, int out_size, void* d_ws, size_t ws_size,
                              hipStream_t stream) {
    (void)in_sizes; (void)n_in; (void)out_size; (void)ws_size;

    const float* x      = (const float*)d_in[0];   // [B,S,D]
    const float* vector = (const float*)d_in[1];   // [B,VDIM]
    const float* W_att  = (const float*)d_in[2];   // [D,H]
    const float* temp   = (const float*)d_in[3];   // [1,1,H]
    const float* W_val  = (const float*)d_in[4];   // [D,H*K]
    const float* W_op   = (const float*)d_in[5];   // [H*K,VDIM]
    const float* ff1    = (const float*)d_in[6];   // [VDIM+D, D]
    const float* ff1_b  = (const float*)d_in[7];   // [1,1,D]
    const float* ff2    = (const float*)d_in[8];   // [D,D]
    const float* ff2_b  = (const float*)d_in[9];   // [1,1,D]
    const float* gamma  = (const float*)d_in[10];  // [D]
    const float* beta   = (const float*)d_in[11];  // [D]
    float* out = (float*)d_out;                    // [B,S,D]

    // workspace layout
    char*  base = (char*)d_ws;
    size_t off  = 0;
    auto alloc = [&](size_t bytes) -> char* {
        char* p = base + off;
        off += (bytes + 255) & ~(size_t)255;
        return p;
    };
    unsigned short* concat  = (unsigned short*)alloc((size_t)MM * CONCATD * 2); // 64MB
    unsigned short* wt      = (unsigned short*)alloc((size_t)DD * CONCATD * 2); // 16MB (reused)
    float*          vbuf    = (float*)alloc((size_t)MM * DD * 4);               // 64MB
    unsigned short* pooled  = (unsigned short*)alloc((size_t)MM * DD * 2);      // 32MB
    unsigned short* h1      = (unsigned short*)alloc((size_t)MM * DD * 2);      // 32MB
    float*          logits  = (float*)alloc((size_t)MM * HH * 4);
    float*          rowmax  = (float*)alloc((size_t)BB * HH * 4);
    float*          coef    = (float*)alloc((size_t)MM * HH * 4);
    float*          partial = (float*)alloc((size_t)BB * NCHUNK * DD * 4);      // 512KB

    const dim3 blk(256);
    const dim3 gemm_grid(DD / 128, MM / 128);  // (16, 64)

    // 1) x -> bf16 into concat[:, 0:D]
    k_x_to_concat<<<(MM * (size_t)DD + 255) / 256, blk, 0, stream>>>(x, concat);

    // 2) attention gate coefficients
    k_logits<<<(MM * HH + 255) / 256, blk, 0, stream>>>(x, W_att, logits);
    k_rowmax<<<BB * HH, blk, 0, stream>>>(logits, rowmax);
    k_coef<<<(MM * HH + 255) / 256, blk, 0, stream>>>(logits, rowmax, temp, coef);

    // 3) v = x @ W_values   (f32 out)
    k_cvt_transpose<<<(DD / 64) * (DD / 64), blk, 0, stream>>>(W_val, wt, DD, DD);
    gemm_bf16_wmma<EPI_F32><<<gemm_grid, blk, 0, stream>>>(
        concat, CONCATD, wt, vbuf, DD, DD, nullptr, 0, nullptr, 0, SS);

    // 4) pooled = coef * cumsum_s(v)  -> bf16 (chunked parallel scan)
    k_scan_part<<<(BB * NCHUNK * DD + 255) / 256, blk, 0, stream>>>(vbuf, partial);
    k_scan_offsets<<<(BB * DD + 255) / 256, blk, 0, stream>>>(partial);
    k_scan_apply<<<(BB * NCHUNK * DD + 255) / 256, blk, 0, stream>>>(vbuf, partial, coef, pooled);

    // 5) op = pooled @ W_op + vector[b]  -> bf16 into concat[:, D:2D]
    k_cvt_transpose<<<(DD / 64) * (DD / 64), blk, 0, stream>>>(W_op, wt, DD, DD);
    gemm_bf16_wmma<EPI_BVEC_BF16><<<gemm_grid, blk, 0, stream>>>(
        pooled, DD, wt, concat + DD, CONCATD, DD, vector, DD, nullptr, 0, SS);

    // 6) h1 = relu(concat @ ff1 + ff1_b)  (K = 4096) -> bf16
    k_cvt_transpose<<<(CONCATD / 64) * (DD / 64), blk, 0, stream>>>(ff1, wt, CONCATD, DD);
    gemm_bf16_wmma<EPI_BRELU_BF16><<<gemm_grid, blk, 0, stream>>>(
        concat, CONCATD, wt, h1, DD, CONCATD, ff1_b, 0, nullptr, 0, SS);

    // 7) res = relu(h1 @ ff2 + ff2_b) + x  -> f32 into d_out
    k_cvt_transpose<<<(DD / 64) * (DD / 64), blk, 0, stream>>>(ff2, wt, DD, DD);
    gemm_bf16_wmma<EPI_BRELU_RES><<<gemm_grid, blk, 0, stream>>>(
        h1, DD, wt, out, DD, DD, ff2_b, 0, x, DD, SS);

    // 8) layernorm in place on d_out
    k_layernorm<<<MM, blk, 0, stream>>>(out, gamma, beta);
}
